// GroupedQueryAttention_35570919145592
// MI455X (gfx1250) — matrix operand order repaired and checked by reading, not compiled, not run on hardware
//
#include <hip/hip_runtime.h>

// Problem constants (match reference)
#define BB   2
#define TT   2048
#define CC   2048
#define HH   32
#define HKV  8
#define DH   64
#define GG   4     // H / HKV

typedef __attribute__((ext_vector_type(16))) __bf16 v16bf;
typedef __attribute__((ext_vector_type(8)))  __bf16 v8bf;
typedef __attribute__((ext_vector_type(8)))  float  v8f;
typedef __attribute__((ext_vector_type(4)))  unsigned int u32x4;
typedef __attribute__((ext_vector_type(8)))  int i32x8;
typedef __attribute__((ext_vector_type(4)))  int i32x4;

// gcc-style vector + address-space-qualified pointee types for the async builtin
// (param type revealed by clang diagnostic: 'vector_size(16) int __device__ *')
typedef int v4i __attribute__((vector_size(16)));
typedef __attribute__((address_space(1))) v4i as1_v4i;
typedef __attribute__((address_space(3))) v4i as3_v4i;

union Frag16 { v16bf v; __bf16 h[16]; };

static __device__ __forceinline__ v8f wmma_bf16(v16bf a, v16bf b, v8f c) {
  // D = A(16x32) * B(32x16) + C, f32 accumulate
  return __builtin_amdgcn_wmma_f32_16x16x32_bf16(false, a, false, b, (short)0, c, false, false);
}

// ---------------------------------------------------------------------------
// CDNA5 data-movement helpers (guarded; fall back to synchronous copies)
// ---------------------------------------------------------------------------
static __device__ __forceinline__ unsigned lds_off_of(const void* p) {
  // AS3 offset == low 32 bits of the generic (flat) address of an LDS object
  return (unsigned)reinterpret_cast<unsigned long long>(p);
}

static __device__ __forceinline__ void cp_async16(const void* g, void* l) {
#if __has_builtin(__builtin_amdgcn_global_load_async_to_lds_b128)
  __builtin_amdgcn_global_load_async_to_lds_b128(
      (as1_v4i*)reinterpret_cast<unsigned long long>(g),
      (as3_v4i*)lds_off_of(l),
      0, 0);
#else
  *(v8bf*)l = *(const v8bf*)g;
#endif
}

static __device__ __forceinline__ void async_wait0() {
#if __has_builtin(__builtin_amdgcn_global_load_async_to_lds_b128)
#if __has_builtin(__builtin_amdgcn_s_wait_asynccnt)
  __builtin_amdgcn_s_wait_asynccnt(0);
#else
  asm volatile("s_wait_asynccnt 0x0" ::: "memory");
#endif
#endif
}

static __device__ __forceinline__ void tensor_wait0() {
#if __has_builtin(__builtin_amdgcn_s_wait_tensorcnt)
  __builtin_amdgcn_s_wait_tensorcnt(0);
#else
  asm volatile("s_wait_tensorcnt 0x0" ::: "memory");
#endif
}

#if __has_builtin(__builtin_amdgcn_tensor_load_to_lds)
#define HAVE_TDM 1
// TDM: load a 2D bf16 tile (rows x row_elts) with row stride (elements) into LDS.
static __device__ __forceinline__ void tdm_load_2d_bf16(unsigned lds_byte_off,
                                                        const void* gaddr,
                                                        unsigned rows,
                                                        unsigned row_elts,
                                                        unsigned stride_elts) {
  unsigned long long ga = reinterpret_cast<unsigned long long>(gaddr);
  u32x4 g0;
  g0[0] = 1u;                                           // count=1 (user D#)
  g0[1] = lds_byte_off;                                 // lds_addr
  g0[2] = (unsigned)(ga & 0xffffffffu);                 // global_addr[31:0]
  g0[3] = (unsigned)((ga >> 32) & 0x01ffffffu)          // global_addr[56:32]
        | (2u << 30);                                   // type = 2 ("image")
  i32x8 g1;
  g1[0] = (int)(1u << 16);                              // data_size=1 -> 2 bytes
  g1[1] = (int)((row_elts & 0xffffu) << 16);            // tensor_dim0[15:0] @ bits63:48
  g1[2] = (int)((row_elts >> 16) & 0xffffu)             // tensor_dim0[31:16]
        | (int)((rows & 0xffffu) << 16);                // tensor_dim1[15:0]
  g1[3] = (int)((rows >> 16) & 0xffffu)                 // tensor_dim1[31:16]
        | (int)((row_elts & 0xffffu) << 16);            // tile_dim0
  g1[4] = (int)(rows & 0xffffu);                        // tile_dim1 (tile_dim2=0)
  g1[5] = (int)stride_elts;                             // tensor_dim0_stride[31:0]
  g1[6] = 0;                                            // stride0 hi, stride1 lo
  g1[7] = 0;
  i32x4 z4 = {0, 0, 0, 0};
#if defined(__clang_major__) && (__clang_major__ >= 23)
  i32x8 z8 = {0, 0, 0, 0, 0, 0, 0, 0};
  __builtin_amdgcn_tensor_load_to_lds(g0, g1, z4, z4, z8, 0);
#else
  __builtin_amdgcn_tensor_load_to_lds(g0, g1, z4, z4, 0);
#endif
}
#else
#define HAVE_TDM 0
#endif

// ---------------------------------------------------------------------------
// Kernel 1: f32 -> bf16 conversion
// ---------------------------------------------------------------------------
__global__ __launch_bounds__(256) void f2bf_kernel(const float* __restrict__ in,
                                                   __bf16* __restrict__ out, int n) {
  int i = blockIdx.x * 256 + threadIdx.x;
  if (i < n) out[i] = (__bf16)in[i];
}

// ---------------------------------------------------------------------------
// Kernel 2: bf16 GEMM  C(MxN) = A(MxK) * Bt(NxK)^T, fp32 out.
// Block tile 128x128, k-step 32, 256 threads = 8 waves (4x2 wave grid),
// each wave computes 32x64 via 2x4 WMMA tiles. Double-buffered LDS with
// async global->LDS staging (ASYNCcnt) overlapped with WMMA compute.
// ---------------------------------------------------------------------------
__global__ __launch_bounds__(256) void gemm_bf16_kernel(const __bf16* __restrict__ A,
                                                        const __bf16* __restrict__ Bt,
                                                        float* __restrict__ Cf,
                                                        int M, int N, int K) {
  __shared__ __bf16 sA[2][128 * 32];
  __shared__ __bf16 sB[2][128 * 32];

  const int tid  = threadIdx.x;
  const int lane = tid & 31;
  const int wid  = tid >> 5;
  const int bm   = blockIdx.y * 128;
  const int bn   = blockIdx.x * 128;
  const int wm   = (wid >> 1) * 32;   // wave row offset in tile
  const int wn   = (wid & 1) * 64;    // wave col offset in tile
  const int row  = lane & 15;
  const int lg   = lane >> 4;         // lane group (0/1)
  const int kb8  = lg * 16;           // A-frag k base (contiguous layout)

  v8f acc[2][4];
  v8f vzero = {};
#pragma unroll
  for (int mi = 0; mi < 2; mi++)
#pragma unroll
    for (int ni = 0; ni < 4; ni++) acc[mi][ni] = vzero;

  auto stage = [&](int buf, int k0) {
#pragma unroll
    for (int i = tid; i < 512; i += 256) {
      int r = i >> 2, seg = i & 3;                 // 128 rows x 4 segments of 8 halves
      cp_async16(A  + (size_t)(bm + r) * K + k0 + seg * 8, &sA[buf][r * 32 + seg * 8]);
      cp_async16(Bt + (size_t)(bn + r) * K + k0 + seg * 8, &sB[buf][r * 32 + seg * 8]);
    }
  };

  stage(0, 0);
  async_wait0();
  __syncthreads();

  int buf = 0;
  for (int k0 = 0; k0 < K; k0 += 32) {
    if (k0 + 32 < K) stage(buf ^ 1, k0 + 32);   // async: overlaps with WMMA below

    Frag16 af[2];
#pragma unroll
    for (int mi = 0; mi < 2; mi++) {
      const __bf16* p = &sA[buf][(wm + mi * 16 + row) * 32 + kb8];
      *(v8bf*)&af[mi].h[0] = *(const v8bf*)p;         // k = kb8 .. kb8+7
      *(v8bf*)&af[mi].h[8] = *(const v8bf*)(p + 8);   // k = kb8+8 .. kb8+15
    }
#pragma unroll
    for (int ni = 0; ni < 4; ni++) {
      Frag16 bfr;
      const __bf16* p = &sB[buf][(wn + ni * 16 + row) * 32 + lg * 16];
      *(v8bf*)&bfr.h[0] = *(const v8bf*)p;            // contiguous 16 k-halves
      *(v8bf*)&bfr.h[8] = *(const v8bf*)(p + 8);
#pragma unroll
      for (int mi = 0; mi < 2; mi++)
        acc[mi][ni] = wmma_bf16(af[mi].v, bfr.v, acc[mi][ni]);
    }
    async_wait0();
    __syncthreads();
    buf ^= 1;
  }

  // C/D layout: element (M = r + 8*lg, N = lane&15) in acc[..][r]
#pragma unroll
  for (int mi = 0; mi < 2; mi++)
#pragma unroll
    for (int ni = 0; ni < 4; ni++)
#pragma unroll
      for (int r = 0; r < 8; r++) {
        int gm = bm + wm + mi * 16 + r + (lg << 3);
        int gn = bn + wn + ni * 16 + (lane & 15);
        Cf[(size_t)gm * N + gn] = acc[mi][ni][r];
      }
}

// ---------------------------------------------------------------------------
// Kernel 3: RoPE + repack.
//  qlin (B,T,C) f32        -> Qo (B,H,T,DH)   bf16 (RoPE applied)
//  kvlin (B,T,2,HKV,DH)    -> Ko (B,HKV,T,DH) bf16 (RoPE), Vo same (no RoPE)
// ---------------------------------------------------------------------------
__global__ __launch_bounds__(256) void rope_pack_kernel(const float* __restrict__ qlin,
                                                        const float* __restrict__ kvlin,
                                                        const float* __restrict__ cosT,
                                                        const float* __restrict__ sinT,
                                                        __bf16* __restrict__ Qo,
                                                        __bf16* __restrict__ Ko,
                                                        __bf16* __restrict__ Vo, int n) {
  int idx = blockIdx.x * 256 + threadIdx.x;
  if (idx >= n) return;
  const int i  = idx & 31;
  int r1 = idx >> 5;
  const int head = r1 % (HH + HKV);
  int r2 = r1 / (HH + HKV);
  const int t = r2 % TT;
  const int b = r2 / TT;

  const float c = cosT[t * 32 + i];
  const float s = sinT[t * 32 + i];

  if (head < HH) {
    const float* q = qlin + ((size_t)(b * TT + t)) * CC + head * DH;
    float v0 = q[i], v1 = q[i + 32];
    __bf16* o = Qo + (((size_t)(b * HH + head)) * TT + t) * DH;
    o[i]      = (__bf16)(v0 * c - v1 * s);
    o[i + 32] = (__bf16)(v1 * c + v0 * s);
  } else {
    const int hh = head - HH;
    const float* kv = kvlin + ((size_t)(b * TT + t)) * (2 * HKV * DH);
    const float* k = kv + hh * DH;                  // slot 0
    const float* v = kv + HKV * DH + hh * DH;       // slot 1
    float k0 = k[i], k1 = k[i + 32];
    __bf16* ok = Ko + (((size_t)(b * HKV + hh)) * TT + t) * DH;
    __bf16* ov = Vo + (((size_t)(b * HKV + hh)) * TT + t) * DH;
    ok[i]      = (__bf16)(k0 * c - k1 * s);
    ok[i + 32] = (__bf16)(k1 * c + k0 * s);
    ov[i]      = (__bf16)v[i];
    ov[i + 32] = (__bf16)v[i + 32];
  }
}

// ---------------------------------------------------------------------------
// Kernel 4: causal flash attention (GQA).
// grid = (T/64, H, B), block = 128 (4 waves). Each wave owns a 16-query tile,
// iterates 32-key blocks with online softmax. K-fragments read directly from
// global (contiguous per lane, prefetched); V tile staged in LDS by the
// Tensor Data Mover (TENSORcnt); P re-striped via LDS.
// ---------------------------------------------------------------------------
__global__ __launch_bounds__(128) void flash_attn_kernel(const __bf16* __restrict__ Q,
                                                         const __bf16* __restrict__ Kk,
                                                         const __bf16* __restrict__ V,
                                                         __bf16* __restrict__ O) {
  const int lane = threadIdx.x & 31;
  const int wid  = threadIdx.x >> 5;
  const int qblk = blockIdx.x;
  const int head = blockIdx.y;
  const int b    = blockIdx.z;
  const int kvh  = head / GG;
  const int q0   = qblk * 64 + wid * 16;

  const __bf16* Qh = Q  + (((size_t)(b * HH  + head)) * TT + q0) * DH;
  const __bf16* Kh = Kk +  ((size_t)(b * HKV + kvh))  * TT * DH;
  const __bf16* Vh = V  +  ((size_t)(b * HKV + kvh))  * TT * DH;

  __shared__ __bf16 sV[32 * 64];        // 32 keys x 64 dims
  __shared__ __bf16 sP[4][16 * 32];     // per-wave P tile (16q x 32k)

  const int row = lane & 15;
  const int lg  = lane >> 4;
  const int kb8 = lg * 16;

  // Q A-fragments for the two 32-wide d-steps
  Frag16 qf[2];
#pragma unroll
  for (int s = 0; s < 2; s++) {
    const __bf16* p = Qh + row * DH + s * 32 + kb8;
    *(v8bf*)&qf[s].h[0] = *(const v8bf*)p;
    *(v8bf*)&qf[s].h[8] = *(const v8bf*)(p + 8);
  }

  float m_i[8], l_i[8];
  v8f o[4];
  v8f vzero = {};
#pragma unroll
  for (int r = 0; r < 8; r++) { m_i[r] = -1e30f; l_i[r] = 0.0f; }
#pragma unroll
  for (int d = 0; d < 4; d++) o[d] = vzero;

  const float scale = 0.125f;            // 1/sqrt(64)
  const int myLimit  = q0 + 15;          // last key this wave may attend to
  const int blkLimit = qblk * 64 + 63;   // uniform loop bound for the block

  for (int kb = 0; kb <= blkLimit; kb += 32) {
    // --- stage V[kb..kb+31][0..63] into LDS
#if HAVE_TDM
    if (wid == 0) {
      tdm_load_2d_bf16(lds_off_of(sV), Vh + (size_t)kb * DH,
                       /*rows=*/32, /*row_elts=*/DH, /*stride_elts=*/DH);
      tensor_wait0();
    }
#else
    {
      int r = threadIdx.x >> 2, seg = threadIdx.x & 3;
      const __bf16* g = Vh + (size_t)(kb + r) * DH + seg * 16;
      __bf16* l = sV + r * 64 + seg * 16;
      *(v8bf*)l       = *(const v8bf*)g;
      *(v8bf*)(l + 8) = *(const v8bf*)(g + 8);
    }
#endif
    // prefetch next K block into cache while we wait/compute
    if (kb + 32 <= blkLimit)
      __builtin_prefetch(Kh + (size_t)(kb + 32 + (lane & 31)) * DH, 0, 0);
    __syncthreads();

    if (kb <= myLimit) {
      // --- S = Q * K^T over 32 keys (2 n-tiles x 2 d-steps)
      v8f sfr[2];
      sfr[0] = vzero; sfr[1] = vzero;
#pragma unroll
      for (int nt = 0; nt < 2; nt++) {
#pragma unroll
        for (int s = 0; s < 2; s++) {
          Frag16 kf;  // B-operand: col = key (lane&15), reduction dim = d (contiguous)
          const __bf16* kp = Kh + (size_t)(kb + nt * 16 + row) * DH + s * 32 + lg * 16;
          *(v8bf*)&kf.h[0] = *(const v8bf*)kp;
          *(v8bf*)&kf.h[8] = *(const v8bf*)(kp + 8);
          sfr[nt] = wmma_bf16(qf[s].v, kf.v, sfr[nt]);
        }
      }

      // --- online softmax (rows = r + 8*lg, cols across 16-lane group)
      float alpha[8];
#pragma unroll
      for (int r = 0; r < 8; r++) {
        const int qrow = q0 + r + (lg << 3);
        const int k0g  = kb + (lane & 15);
        float s0 = sfr[0][r] * scale;
        float s1 = sfr[1][r] * scale;
        if (k0g      > qrow) s0 = -1e30f;
        if (k0g + 16 > qrow) s1 = -1e30f;
        float mt = fmaxf(s0, s1);
#pragma unroll
        for (int off = 1; off < 16; off <<= 1)
          mt = fmaxf(mt, __shfl_xor(mt, off, 32));
        float mnew = fmaxf(m_i[r], mt);
        float p0 = __expf(s0 - mnew);
        float p1 = __expf(s1 - mnew);
        float rs = p0 + p1;
#pragma unroll
        for (int off = 1; off < 16; off <<= 1)
          rs += __shfl_xor(rs, off, 32);
        alpha[r] = __expf(m_i[r] - mnew);
        l_i[r]   = l_i[r] * alpha[r] + rs;
        m_i[r]   = mnew;
        __bf16* pp = sP[wid] + (r + (lg << 3)) * 32;
        pp[lane & 15]        = (__bf16)p0;
        pp[(lane & 15) + 16] = (__bf16)p1;
      }
#pragma unroll
      for (int d = 0; d < 4; d++)
#pragma unroll
        for (int r = 0; r < 8; r++) o[d][r] *= alpha[r];

      // --- reload P in A-fragment striping (same-wave DS ordering is safe)
      Frag16 pf;
      {
        const __bf16* p = sP[wid] + row * 32 + kb8;
        *(v8bf*)&pf.h[0] = *(const v8bf*)p;
        *(v8bf*)&pf.h[8] = *(const v8bf*)(p + 8);
      }
      // --- O += P * V (4 d-chunks of 16)
#pragma unroll
      for (int d = 0; d < 4; d++) {
        Frag16 vf;  // B-operand: col = d-chunk element, reduction dim = key
#pragma unroll
        for (int h = 0; h < 16; h++)
          vf.h[h] = sV[((lg << 4) + h) * 64 + d * 16 + (lane & 15)];
        o[d] = wmma_bf16(pf.v, vf.v, o[d]);
      }
    }
    __syncthreads();   // sV reused next iteration
  }

  // --- epilogue: normalize and scatter to attn (B,T,H*DH) bf16
#pragma unroll
  for (int r = 0; r < 8; r++) l_i[r] = 1.0f / l_i[r];
#pragma unroll
  for (int d = 0; d < 4; d++)
#pragma unroll
    for (int r = 0; r < 8; r++) {
      int t = q0 + r + (lg << 3);
      O[((size_t)b * TT + t) * CC + head * DH + d * 16 + (lane & 15)] =
          (__bf16)(o[d][r] * l_i[r]);
    }
}

// ---------------------------------------------------------------------------
// Host launcher
// ---------------------------------------------------------------------------
extern "C" void kernel_launch(void* const* d_in, const int* in_sizes, int n_in,
                              void* d_out, int out_size, void* d_ws, size_t ws_size,
                              hipStream_t stream) {
  const float* x    = (const float*)d_in[0];
  const float* cosT = (const float*)d_in[1];
  const float* sinT = (const float*)d_in[2];
  const float* Wq   = (const float*)d_in[3];
  const float* Wkv  = (const float*)d_in[4];
  const float* Wo   = (const float*)d_in[5];
  float* out = (float*)d_out;

  char* ws = (char*)d_ws;
  const size_t N_X   = (size_t)BB * TT * CC;        // 8,388,608
  const size_t N_WQ  = (size_t)CC * CC;             // 4,194,304
  const size_t N_WKV = (size_t)(2 * HKV * DH) * CC; // 2,097,152
  const size_t N_KV  = (size_t)BB * TT * (2 * HKV * DH);
  const size_t N_HD  = (size_t)BB * HKV * TT * DH;  // per kv tensor

  size_t off = 0;
  __bf16* xb    = (__bf16*)(ws + off); off += N_X   * 2;
  __bf16* Wqb   = (__bf16*)(ws + off); off += N_WQ  * 2;
  __bf16* Wkvb  = (__bf16*)(ws + off); off += N_WKV * 2;
  __bf16* Wob   = (__bf16*)(ws + off); off += N_WQ  * 2;
  float*  qlin  = (float*) (ws + off); off += N_X   * 4;
  float*  kvlin = (float*) (ws + off); off += N_KV  * 4;
  __bf16* Qr    = (__bf16*)(ws + off); off += N_X   * 2;   // (B,H,T,DH)
  __bf16* Kr    = (__bf16*)(ws + off); off += N_HD  * 2;
  __bf16* Vr    = (__bf16*)(ws + off); off += N_HD  * 2;
  __bf16* attn  = (__bf16*)(ws + off); off += N_X   * 2;   // (B,T,C)
  (void)ws_size; (void)in_sizes; (void)n_in; (void)out_size;

  // 1) f32 -> bf16 conversions
  f2bf_kernel<<<(N_X   + 255) / 256, 256, 0, stream>>>(x,   xb,   (int)N_X);
  f2bf_kernel<<<(N_WQ  + 255) / 256, 256, 0, stream>>>(Wq,  Wqb,  (int)N_WQ);
  f2bf_kernel<<<(N_WKV + 255) / 256, 256, 0, stream>>>(Wkv, Wkvb, (int)N_WKV);
  f2bf_kernel<<<(N_WQ  + 255) / 256, 256, 0, stream>>>(Wo,  Wob,  (int)N_WQ);

  const int M = BB * TT;  // 4096
  // 2) Q = x @ Wq^T  (M x 2048 x 2048)
  gemm_bf16_kernel<<<dim3(CC / 128, M / 128), 256, 0, stream>>>(xb, Wqb, qlin, M, CC, CC);
  // 3) KV = x @ Wkv^T (M x 1024 x 2048)
  gemm_bf16_kernel<<<dim3((2 * HKV * DH) / 128, M / 128), 256, 0, stream>>>(
      xb, Wkvb, kvlin, M, 2 * HKV * DH, CC);

  // 4) RoPE + repack
  {
    const int n = BB * TT * (HH + HKV) * 32;
    rope_pack_kernel<<<(n + 255) / 256, 256, 0, stream>>>(qlin, kvlin, cosT, sinT,
                                                          Qr, Kr, Vr, n);
  }

  // 5) causal flash attention
  flash_attn_kernel<<<dim3(TT / 64, HH, BB), 128, 0, stream>>>(Qr, Kr, Vr, attn);

  // 6) out = attn @ Wo^T -> fp32 d_out
  gemm_bf16_kernel<<<dim3(CC / 128, M / 128), 256, 0, stream>>>(attn, Wob, out, M, CC, CC);
}
